// GNN_44942537785535
// MI455X (gfx1250) — matrix operand order repaired and checked by reading, not compiled, not run on hardware
//
#include <hip/hip_runtime.h>

typedef __bf16 bf16_t;
typedef __attribute__((ext_vector_type(16))) __bf16 v16bf;
typedef __attribute__((ext_vector_type(8)))  float  v8f;
typedef __attribute__((ext_vector_type(4)))  unsigned int u32x4;
typedef __attribute__((ext_vector_type(8)))  int i32x8;
typedef __attribute__((ext_vector_type(4)))  int i32x4;

#define NN  4096
#define DD  512
#define HH  8
#define DHH 64
#define EE  131072
#define LL  2

#if defined(__gfx1250__)
#if __has_builtin(__builtin_amdgcn_tensor_load_to_lds)
#define USE_TDM 1
#else
#error "gfx1250 device pass: __builtin_amdgcn_tensor_load_to_lds unavailable"
#endif
#endif

// ---------------------------------------------------------------- WMMA helper
__device__ __forceinline__ v8f wmma_bf16(v16bf a, v16bf b, v8f c) {
  // (neg_a, A, neg_b, B, c_mod, C, reuse_a, reuse_b)
  return __builtin_amdgcn_wmma_f32_16x16x32_bf16(false, a, false, b, (short)0, c,
                                                 false, false);
}

// wave32 cross-lane reductions within each 16-lane half (matches WMMA C/D layout)
__device__ __forceinline__ float redmax16(float v) {
  v = fmaxf(v, __shfl_xor(v, 1));
  v = fmaxf(v, __shfl_xor(v, 2));
  v = fmaxf(v, __shfl_xor(v, 4));
  v = fmaxf(v, __shfl_xor(v, 8));
  return v;
}
__device__ __forceinline__ float redsum16(float v) {
  v += __shfl_xor(v, 1);
  v += __shfl_xor(v, 2);
  v += __shfl_xor(v, 4);
  v += __shfl_xor(v, 8);
  return v;
}

// pack 4 f32 -> 4 bf16 and store as one 8-byte LDS write (dst must be 8B aligned)
__device__ __forceinline__ void st_bf16x4(bf16_t* __restrict__ d, float a, float b,
                                          float c, float e) {
  union {
    bf16_t h[4];
    uint2 u;
  } t;
  t.h[0] = (bf16_t)a;
  t.h[1] = (bf16_t)b;
  t.h[2] = (bf16_t)c;
  t.h[3] = (bf16_t)e;
  *reinterpret_cast<uint2*>(d) = t.u;
}

// ------------------------------------------------------- TDM 2D panel loader
#ifdef USE_TDM
// Loads a [DHH x 128] bf16 tile out of a [DHH x NN] row-major tensor into LDS.
__device__ __forceinline__ void tdm_load2d_bf16(const bf16_t* g, unsigned lds_off) {
  unsigned long long ga = (unsigned long long)(size_t)g;
  u32x4 g0;
  g0[0] = 1u;                                                // count=1, user mode
  g0[1] = lds_off;                                           // lds_addr
  g0[2] = (unsigned)ga;                                      // global_addr[31:0]
  g0[3] = (unsigned)((ga >> 32) & 0x01ffffffu) | (2u << 30); // addr hi | type=2
  i32x8 g1;
  g1[0] = (int)(1u << 16);                                   // wg_mask=0, 2B elems
  g1[1] = (int)(((unsigned)NN & 0xffffu) << 16);             // tensor_dim0 lo
  g1[2] = (int)((((unsigned)NN >> 16) & 0xffffu) | (((unsigned)DHH & 0xffffu) << 16));
  g1[3] = (int)(((unsigned)DHH >> 16) | (128u << 16));       // tile_dim0 = 128
  g1[4] = (int)((unsigned)DHH & 0xffffu);                    // tile_dim1=64
  g1[5] = (int)NN;                                           // tensor_dim0_stride
  g1[6] = 0;
  g1[7] = 0;
  i32x4 z4 = {0, 0, 0, 0};
#if __clang_major__ >= 23
  i32x8 z8 = {0, 0, 0, 0, 0, 0, 0, 0};
  __builtin_amdgcn_tensor_load_to_lds(g0, g1, z4, z4, z8, 0);
#else
  __builtin_amdgcn_tensor_load_to_lds(g0, g1, z4, z4, 0);
#endif
}
#endif

// ------------------------------------------------------------------- GEMM
// C[M=4096, N=512] = A[4096,512] @ B (B = [K,N] or, if TRANSB, B[n*K+k])
// f32 inputs converted to bf16 while staging to LDS; f32 accumulate.
// B is staged transposed (BsT[n][k]) so B-fragments are contiguous ds_load_b128s.
template <bool TRANSB>
__global__ __launch_bounds__(256) void gemm_bf16(const float* __restrict__ A,
                                                 const float* __restrict__ B,
                                                 const float* __restrict__ bias,
                                                 float scale,
                                                 float* __restrict__ Cf,
                                                 bf16_t* __restrict__ Cb) {
  __shared__ bf16_t As[64][40];    // rows 80 B (16B aligned)
  __shared__ bf16_t BsT[128][40];  // transposed B tile, rows 80 B
  const int tid = threadIdx.x;
  const int lane = tid & 31, w = tid >> 5;
  const int wr = w >> 2, wc = w & 3;  // 2x4 waves -> block tile 64x128
  const int m0 = blockIdx.x * 64;
  const int n0 = blockIdx.y * 128;
  const int lm = lane & 15, kh = lane >> 4;
  v8f acc[2][2] = {};

  for (int kc = 0; kc < DD; kc += 32) {
    if (kc + 32 < DD)  // global_prefetch_b8 of next A panel
      __builtin_prefetch(A + (size_t)(m0 + (tid >> 2)) * DD + kc + 32, 0, 1);
    // stage A: 64x32, float4 reads, packed bf16x4 LDS writes
    for (int t = tid; t < 512; t += 256) {
      int idx = t * 4;
      int r = idx >> 5, c = idx & 31;
      const float4 v = *(const float4*)(A + (size_t)(m0 + r) * DD + kc + c);
      st_bf16x4(&As[r][c], v.x, v.y, v.z, v.w);
    }
    // stage B transposed: 32x128
    if (TRANSB) {
      for (int t = tid; t < 1024; t += 256) {
        int idx = t * 4;
        int r = idx & 31, c = idx >> 5;  // r (k) fastest -> contiguous reads
        const float4 v = *(const float4*)(B + (size_t)(n0 + c) * DD + kc + r);
        st_bf16x4(&BsT[c][r], v.x, v.y, v.z, v.w);
      }
    } else {
      for (int t = tid; t < 1024; t += 256) {
        int idx = t * 4;
        int c = idx & 127, r = idx >> 7;  // c fastest -> contiguous reads
        const float4 v = *(const float4*)(B + (size_t)(kc + r) * DD + n0 + c);
        BsT[c + 0][r] = (bf16_t)v.x;
        BsT[c + 1][r] = (bf16_t)v.y;
        BsT[c + 2][r] = (bf16_t)v.z;
        BsT[c + 3][r] = (bf16_t)v.w;
      }
    }
    __syncthreads();

    v16bf af[2], bfr[2];
#pragma unroll
    for (int mt = 0; mt < 2; ++mt) {
      int ar = wr * 32 + mt * 16 + lm;
#pragma unroll
      for (int e = 0; e < 16; ++e) {
        int k = ((e >> 3) << 4) + kh * 8 + (e & 7);  // two contiguous 16B runs
        af[mt][e] = As[ar][k];
      }
    }
#pragma unroll
    for (int nt = 0; nt < 2; ++nt) {
      int bc = wc * 32 + nt * 16 + lm;
#pragma unroll
      for (int e = 0; e < 16; ++e) bfr[nt][e] = BsT[bc][((e >> 3) << 4) + kh * 8 + (e & 7)];  // 2x b128
    }
#pragma unroll
    for (int mt = 0; mt < 2; ++mt)
#pragma unroll
      for (int nt = 0; nt < 2; ++nt)
        acc[mt][nt] = wmma_bf16(af[mt], bfr[nt], acc[mt][nt]);
    __syncthreads();
  }

  const int rbase = kh << 3;  // lanes 16..31 own rows 8..15
#pragma unroll
  for (int mt = 0; mt < 2; ++mt)
#pragma unroll
    for (int nt = 0; nt < 2; ++nt) {
      int col = n0 + wc * 32 + nt * 16 + lm;
      float bv = bias ? bias[col] : 0.f;
#pragma unroll
      for (int j = 0; j < 8; ++j) {
        int row = m0 + wr * 32 + mt * 16 + rbase + j;
        float v = (acc[mt][nt][j] + bv) * scale;
        if (Cf) Cf[(size_t)row * DD + col] = v;
        if (Cb) Cb[(size_t)row * DD + col] = (bf16_t)v;
      }
    }
}

// --------------------------------------------------------- flash attention
// Faithful to the reference's raw reshape: Qh[d][n] = Q_flat[h*DH*N + d*N + n].
// Output a[h][n][d] stored at h*N*DH + n*DH + d (== a.reshape(N,D) row-major).
// K panel staged TRANSPOSED in LDS (contiguous score B-frags); V panel staged
// raw by the Tensor Data Mover; P staged as bf16 for the PV A-frags.
__global__ __launch_bounds__(256) void attn_fa(const bf16_t* __restrict__ Q,
                                               const bf16_t* __restrict__ K,
                                               const bf16_t* __restrict__ V,
                                               float* __restrict__ Aout) {
  __shared__ bf16_t Kswz[128][72];  // [key m][d], rows 144 B (16B aligned)
  __shared__ bf16_t Vs[DHH][128];   // raw [d][m] (TDM destination)
  __shared__ bf16_t Psb[8][16][40]; // per-wave P tiles, rows 80 B
  const int tid = threadIdx.x;
  const int lane = tid & 31, w = tid >> 5;
  const int lm = lane & 15, kh = lane >> 4;
  const int head = blockIdx.y;
  const int nq = blockIdx.x * 128 + w * 16;  // this wave's 16 query rows
  const size_t hb = (size_t)head * DHH * NN;

  // Q A-fragments (16 rows x 64 k), resident for the whole kernel
  v16bf aq[2];
#pragma unroll
  for (int f = 0; f < 2; ++f)
#pragma unroll
    for (int e = 0; e < 16; ++e) {
      int d = f * 32 + kh * 16 + e;
      aq[f][e] = Q[hb + (size_t)d * NN + nq + lm];
    }

  v8f O[4] = {};
  float rmax[8], rsum[8];
#pragma unroll
  for (int j = 0; j < 8; ++j) {
    rmax[j] = -3.0e38f;
    rsum[j] = 0.f;
  }

  for (int p0 = 0; p0 < NN; p0 += 128) {
    __syncthreads();  // previous panel fully consumed
#ifdef USE_TDM
    if (tid < 32)  // wave 0 drives the Tensor Data Mover for the V panel
      tdm_load2d_bf16(V + hb + p0, (unsigned)(size_t)&Vs[0][0]);
#endif
    // cooperative transposed staging of the K panel (uint4 global reads)
    for (int t = tid; t < 1024; t += 256) {
      int d = t >> 4, mb = (t & 15) * 8;
      union {
        uint4 u;
        bf16_t h[8];
      } r;
      r.u = *(const uint4*)(K + hb + (size_t)d * NN + p0 + mb);
#pragma unroll
      for (int i = 0; i < 8; ++i) Kswz[mb + i][d] = r.h[i];
    }
#ifdef USE_TDM
    if (tid < 32) __builtin_amdgcn_s_wait_tensorcnt(0);
#else
    for (int t = tid; t < 1024; t += 256) {
      int d = t >> 4, mb = (t & 15) * 8;
      *(uint4*)&Vs[d][mb] = *(const uint4*)(V + hb + (size_t)d * NN + p0 + mb);
    }
#endif
    __syncthreads();

    for (int mc = 0; mc < 128; mc += 32) {
      v8f S0 = {}, S1 = {};
      {
        v16bf b0, b1;
#pragma unroll
        for (int e = 0; e < 16; ++e) {  // contiguous 32B runs per half
          b0[e] = Kswz[mc + lm][kh * 16 + e];
          b1[e] = Kswz[mc + lm][32 + kh * 16 + e];
        }
        S0 = wmma_bf16(aq[0], b0, S0);
        S0 = wmma_bf16(aq[1], b1, S0);
#pragma unroll
        for (int e = 0; e < 16; ++e) {
          b0[e] = Kswz[mc + 16 + lm][kh * 16 + e];
          b1[e] = Kswz[mc + 16 + lm][32 + kh * 16 + e];
        }
        S1 = wmma_bf16(aq[0], b0, S1);
        S1 = wmma_bf16(aq[1], b1, S1);
      }
      // online softmax over this 32-key chunk (per query row)
      float fac[8];
#pragma unroll
      for (int j = 0; j < 8; ++j) {
        float tm = redmax16(fmaxf(S0[j], S1[j]));
        float nm = fmaxf(rmax[j], tm);
        float f0 = __expf(rmax[j] - nm);
        float e0 = __expf(S0[j] - nm);
        float e1 = __expf(S1[j] - nm);
        rsum[j] = rsum[j] * f0 + redsum16(e0 + e1);
        rmax[j] = nm;
        fac[j] = f0;
        int r = (kh << 3) + j;  // C-layout row owned by this lane
        Psb[w][r][lm] = (bf16_t)e0;
        Psb[w][r][lm + 16] = (bf16_t)e1;
      }
#pragma unroll
      for (int t = 0; t < 4; ++t)
#pragma unroll
        for (int j = 0; j < 8; ++j) O[t][j] *= fac[j];

      // P (C-layout) -> A-fragment via per-wave LDS, then PV.
      v16bf ap;
#pragma unroll
      for (int e = 0; e < 16; ++e) {
        int k = kh * 16 + e;
        ap[e] = Psb[w][lm][k];
      }
#pragma unroll
      for (int t = 0; t < 4; ++t) {
        v16bf bv;
#pragma unroll
        for (int e = 0; e < 16; ++e) bv[e] = Vs[t * 16 + lm][mc + kh * 16 + e];
        O[t] = wmma_bf16(ap, bv, O[t]);
      }
    }
  }

#pragma unroll
  for (int t = 0; t < 4; ++t)
#pragma unroll
    for (int j = 0; j < 8; ++j) {
      int row = (kh << 3) + j;
      Aout[(size_t)head * NN * DHH + (size_t)(nq + row) * DHH + t * 16 + lm] =
          O[t][j] / rsum[j];
    }
}

// --------------------------------------------------------------- small kernels
__global__ void kzero(float* __restrict__ p, int n) {
  int i = blockIdx.x * 256 + threadIdx.x;
  if (i < n) p[i] = 0.f;
}

__global__ void kdeg(const int* __restrict__ ei, float* __restrict__ deg) {
  int e = blockIdx.x * 256 + threadIdx.x;
  if (e < EE) atomicAdd(&deg[ei[EE + e]], 1.0f);
}

__global__ void kdinv(const float* __restrict__ deg, float* __restrict__ dinv) {
  int i = blockIdx.x * 256 + threadIdx.x;
  if (i < NN) {
    float d = deg[i];
    dinv[i] = d > 0.f ? rsqrtf(d) : 0.f;
  }
}

__global__ void kew(const int* __restrict__ ei, const float* __restrict__ dinv,
                    float* __restrict__ ew) {
  int e = blockIdx.x * 256 + threadIdx.x;
  if (e < EE) ew[e] = dinv[ei[e]] * dinv[ei[EE + e]];
}

__global__ void kscatter(const int* __restrict__ ei, const float* __restrict__ ew,
                         const float* __restrict__ h, float* __restrict__ g) {
  long long idx = (long long)blockIdx.x * 256 + threadIdx.x;  // EE * (DD/4)
  int e = (int)(idx >> 7);
  int c = ((int)idx & 127) << 2;
  if (e >= EE) return;
  const float4 v = *(const float4*)(h + (size_t)ei[e] * DD + c);
  float wgt = ew[e];
  float* dst = g + (size_t)ei[EE + e] * DD + c;
  atomicAdd(dst + 0, v.x * wgt);
  atomicAdd(dst + 1, v.y * wgt);
  atomicAdd(dst + 2, v.z * wgt);
  atomicAdd(dst + 3, v.w * wgt);
}

__global__ __launch_bounds__(256) void kcolsum(const float* __restrict__ g,
                                               float* __restrict__ red) {
  int r0 = blockIdx.x * 32;
  int c0 = threadIdx.x * 2;
  float a0 = 0.f, a1 = 0.f;
  for (int r = 0; r < 32; ++r) {
    a0 += g[(size_t)(r0 + r) * DD + c0];
    a1 += g[(size_t)(r0 + r) * DD + c0 + 1];
  }
  atomicAdd(&red[c0], a0);
  atomicAdd(&red[c0 + 1], a1);
}

__global__ __launch_bounds__(256) void kcenter(const float* __restrict__ g,
                                               const float* __restrict__ red,
                                               float* __restrict__ h,
                                               float* __restrict__ sq) {
  __shared__ float sb[256];
  int row = blockIdx.x, t = threadIdx.x;
  float c0 = g[(size_t)row * DD + t] - red[t] * (1.f / NN);
  float c1 = g[(size_t)row * DD + t + 256] - red[t + 256] * (1.f / NN);
  h[(size_t)row * DD + t] = c0;
  h[(size_t)row * DD + t + 256] = c1;
  sb[t] = c0 * c0 + c1 * c1;
  __syncthreads();
  for (int s = 128; s > 0; s >>= 1) {
    if (t < s) sb[t] += sb[t + s];
    __syncthreads();
  }
  if (t == 0) atomicAdd(sq, sb[0]);
}

// PairNorm finalize + (relu(h)+h) + LayerNorm1, in place on h.
__global__ __launch_bounds__(256) void kfinln1(float* __restrict__ h,
                                               const float* __restrict__ sq,
                                               const float* __restrict__ gam,
                                               const float* __restrict__ bet) {
  __shared__ float sb[256];
  int row = blockIdx.x, t = threadIdx.x;
  float s = rsqrtf(1e-5f + sq[0] * (1.f / NN));
  float v0 = h[(size_t)row * DD + t] * s;
  float v1 = h[(size_t)row * DD + t + 256] * s;
  v0 = fmaxf(v0, 0.f) + v0;
  v1 = fmaxf(v1, 0.f) + v1;
  sb[t] = v0 + v1;
  __syncthreads();
  for (int s2 = 128; s2 > 0; s2 >>= 1) {
    if (t < s2) sb[t] += sb[t + s2];
    __syncthreads();
  }
  float mu = sb[0] * (1.f / DD);
  __syncthreads();
  float d0 = v0 - mu, d1 = v1 - mu;
  sb[t] = d0 * d0 + d1 * d1;
  __syncthreads();
  for (int s2 = 128; s2 > 0; s2 >>= 1) {
    if (t < s2) sb[t] += sb[t + s2];
    __syncthreads();
  }
  float inv = rsqrtf(sb[0] * (1.f / DD) + 1e-5f);
  h[(size_t)row * DD + t] = d0 * inv * gam[t] + bet[t];
  h[(size_t)row * DD + t + 256] = d1 * inv * gam[t + 256] + bet[t + 256];
}

__global__ __launch_bounds__(256) void kln(const float* __restrict__ src,
                                           const float* __restrict__ gam,
                                           const float* __restrict__ bet,
                                           float* __restrict__ dst) {
  __shared__ float sb[256];
  int row = blockIdx.x, t = threadIdx.x;
  float v0 = src[(size_t)row * DD + t];
  float v1 = src[(size_t)row * DD + t + 256];
  sb[t] = v0 + v1;
  __syncthreads();
  for (int s2 = 128; s2 > 0; s2 >>= 1) {
    if (t < s2) sb[t] += sb[t + s2];
    __syncthreads();
  }
  float mu = sb[0] * (1.f / DD);
  __syncthreads();
  float d0 = v0 - mu, d1 = v1 - mu;
  sb[t] = d0 * d0 + d1 * d1;
  __syncthreads();
  for (int s2 = 128; s2 > 0; s2 >>= 1) {
    if (t < s2) sb[t] += sb[t + s2];
    __syncthreads();
  }
  float inv = rsqrtf(sb[0] * (1.f / DD) + 1e-5f);
  dst[(size_t)row * DD + t] = d0 * inv * gam[t] + bet[t];
  dst[(size_t)row * DD + t + 256] = d1 * inv * gam[t + 256] + bet[t + 256];
}

// --------------------------------------------------------------------- driver
extern "C" void kernel_launch(void* const* d_in, const int* in_sizes, int n_in,
                              void* d_out, int out_size, void* d_ws, size_t ws_size,
                              hipStream_t stream) {
  (void)in_sizes; (void)n_in; (void)out_size; (void)ws_size;
  const float* x   = (const float*)d_in[0];
  const int*   ei  = (const int*)d_in[1];
  // d_in[2] edge_attr: unused by the reference computation
  const float* Wg  = (const float*)d_in[3];
  const float* l1g = (const float*)d_in[4];
  const float* l1b = (const float*)d_in[5];
  const float* Wq  = (const float*)d_in[6];
  const float* bq  = (const float*)d_in[7];
  const float* Wk  = (const float*)d_in[8];
  const float* bk  = (const float*)d_in[9];
  const float* Wv  = (const float*)d_in[10];
  const float* bv  = (const float*)d_in[11];
  const float* Wo  = (const float*)d_in[12];
  const float* bo  = (const float*)d_in[13];
  const float* l2g = (const float*)d_in[14];
  const float* l2b = (const float*)d_in[15];
  float* out = (float*)d_out;

  char* p = (char*)d_ws;
  auto carve = [&](size_t b) {
    char* r = p;
    p += (b + 255) & ~(size_t)255;
    return r;
  };
  const size_t ND = (size_t)NN * DD;
  float*  hbuf = (float*)carve(ND * 4);
  float*  gbuf = (float*)carve(ND * 4);
  float*  abuf = (float*)carve(ND * 4);
  float*  xcur = (float*)carve(ND * 4);
  bf16_t* qb   = (bf16_t*)carve(ND * 2);
  bf16_t* kb   = (bf16_t*)carve(ND * 2);
  bf16_t* vb   = (bf16_t*)carve(ND * 2);
  float*  deg  = (float*)carve(NN * 4);
  float*  dinv = (float*)carve(NN * 4);
  float*  ew   = (float*)carve(EE * 4);
  float*  red  = (float*)carve(1024 * 4);  // [0..511]=col sums, [512]=sq-sum

  const float scl = 0.044194173824159216f;  // 1/sqrt(512), applied to Q and K

  kzero<<<(NN + 255) / 256, 256, 0, stream>>>(deg, NN);
  kdeg<<<EE / 256, 256, 0, stream>>>(ei, deg);
  kdinv<<<NN / 256, 256, 0, stream>>>(deg, dinv);
  kew<<<EE / 256, 256, 0, stream>>>(ei, dinv, ew);

  dim3 ggrid(NN / 64, DD / 128);
  for (int i = 0; i < LL; ++i) {
    const float* xin = (i == 0) ? x : xcur;
    gemm_bf16<false><<<ggrid, 256, 0, stream>>>(xin, Wg + (size_t)i * DD * DD,
                                                nullptr, 1.f, hbuf, nullptr);
    kzero<<<(int)(ND / 256), 256, 0, stream>>>(gbuf, (int)ND);
    kscatter<<<EE * (DD / 4) / 256, 256, 0, stream>>>(ei, ew, hbuf, gbuf);
    kzero<<<4, 256, 0, stream>>>(red, 1024);
    kcolsum<<<NN / 32, 256, 0, stream>>>(gbuf, red);
    kcenter<<<NN, 256, 0, stream>>>(gbuf, red, hbuf, red + 512);
    kfinln1<<<NN, 256, 0, stream>>>(hbuf, red + 512, l1g + (size_t)i * DD,
                                    l1b + (size_t)i * DD);
    gemm_bf16<true><<<ggrid, 256, 0, stream>>>(hbuf, Wq + (size_t)i * DD * DD,
                                               bq + (size_t)i * DD, scl, nullptr, qb);
    gemm_bf16<true><<<ggrid, 256, 0, stream>>>(hbuf, Wk + (size_t)i * DD * DD,
                                               bk + (size_t)i * DD, scl, nullptr, kb);
    gemm_bf16<true><<<ggrid, 256, 0, stream>>>(hbuf, Wv + (size_t)i * DD * DD,
                                               bv + (size_t)i * DD, 1.f, nullptr, vb);
    attn_fa<<<dim3(NN / 128, HH), 256, 0, stream>>>(qb, kb, vb, abuf);
    gemm_bf16<true><<<ggrid, 256, 0, stream>>>(abuf, Wo + (size_t)i * DD * DD,
                                               bo + (size_t)i * DD, 1.f, gbuf, nullptr);
    kln<<<NN, 256, 0, stream>>>(gbuf, l2g + (size_t)i * DD, l2b + (size_t)i * DD,
                                (i == LL - 1) ? out : xcur);
  }
}
